// GNN_16209206575854
// MI455X (gfx1250) — matrix operand order repaired and checked by reading, not compiled, not run on hardware
//
#include <hip/hip_runtime.h>
#include <hip/hip_bf16.h>

// ---------- POD vector types (safe in unions; HIP's float4 has ctors) ----------
typedef float  f32x4 __attribute__((ext_vector_type(4)));
typedef __bf16 v16bf __attribute__((ext_vector_type(16)));
typedef float  v8f   __attribute__((ext_vector_type(8)));

union FragBF {
    v16bf v;
    f32x4 f[2];   // two 16B halves of the 32B fragment
};

// CDNA5 async global->LDS copy path (ASYNCcnt-tracked), with safe fallback.
#if defined(__has_builtin)
#if __has_builtin(__builtin_amdgcn_global_load_async_to_lds_b128) && \
    __has_builtin(__builtin_amdgcn_s_wait_asynccnt)
#define USE_ASYNC_LDS 1
#endif
#endif

#if defined(USE_ASYNC_LDS)
// Exact parameter type per compiler diagnostic:
//   "__attribute__((__vector_size__(4 * sizeof(int)))) int __device__ *"  (AS1)
typedef int i32x4v __attribute__((vector_size(16)));
typedef __attribute__((address_space(1))) i32x4v* as1_v4p;
typedef __attribute__((address_space(3))) i32x4v* as3_v4p;

static __device__ __forceinline__ void async_cp16(const void* gsrc, void* ldst) {
    __builtin_amdgcn_global_load_async_to_lds_b128(
        (as1_v4p)(unsigned long long)(uintptr_t)gsrc,
        (as3_v4p)(unsigned)(uintptr_t)ldst,
        0, 0);
}
#endif

#define BROW 40   // padded LDS row stride (bf16 elems): 80B -> conflict-free b128 reads

// =====================================================================
// WMMA GEMM:  C[N,Fout] = A[N,K](bf16) * Bt[Fout,K](bf16, pre-transposed)
// Block = 256 threads = 8 waves; each wave computes 32(M) x 64(N) of C.
// B tile (64 cols x 32 k) staged in LDS once per block per K-step
// (shared by all 8 waves), double-buffered via async global->LDS copies.
// 8 independent WMMAs per wave per K-step; A fragments double-buffered.
// =====================================================================
__global__ __launch_bounds__(256) void gemm_bf16_wmma(
    const __bf16* __restrict__ A,   // [N, K] row-major
    const __bf16* __restrict__ Bt,  // [Fout, K] row-major (W transposed)
    float* __restrict__ C,          // [N, Fout]
    int N, int K, int Fout)
{
    __shared__ __attribute__((aligned(16))) __bf16 Bs[2][64 * BROW];

    const int tid  = threadIdx.x;
    const int lane = tid & 31;
    const int wave = tid >> 5;
    const int half = lane >> 4;     // 0: lanes 0-15, 1: lanes 16-31
    const int l16  = lane & 15;

    const int mtile = blockIdx.x * 8 + wave;     // 32-row tile index
    const int n0    = blockIdx.y * 64;           // first output column
    const bool rowok = (mtile * 32 < N);
    const int rbase  = (rowok ? mtile : 0) * 32;

    // ---- B staging: thread t copies 16B: col bn = t/4, k-chunk bc = (t%4)*8
    const int bn = tid >> 2;
    const int bc = (tid & 3) * 8;
    const __bf16* bsrc = Bt + (size_t)(n0 + bn) * K + bc;
    __bf16* bdst0 = &Bs[0][bn * BROW + bc];
    __bf16* bdst1 = &Bs[1][bn * BROW + bc];

    // ---- A row pointers: two 16-row sets per wave ----
    const __bf16* arow0 = A + (size_t)(rbase + l16) * K;
    const __bf16* arow1 = A + (size_t)(rbase + 16 + l16) * K;

    v8f acc00 = {}, acc01 = {}, acc02 = {}, acc03 = {};
    v8f acc10 = {}, acc11 = {}, acc12 = {}, acc13 = {};
    FragBF a0c, a1c, a0n, a1n;

    const int nst = K / 32;

    // stage 0 of B, A fragments 0
    {
#if defined(USE_ASYNC_LDS)
        async_cp16(bsrc, bdst0);
#else
        *(f32x4*)bdst0 = *(const f32x4*)bsrc;
#endif
        const __bf16* pa = arow0 + half * 8;
        a0c.f[0] = *(const f32x4*)(pa);
        a0c.f[1] = *(const f32x4*)(pa + 16);
        pa = arow1 + half * 8;
        a1c.f[0] = *(const f32x4*)(pa);
        a1c.f[1] = *(const f32x4*)(pa + 16);
    }

    for (int i = 0; i < nst; ++i) {
        const int k0 = i * 32;
        const int buf = i & 1;

        if (i + 1 < nst) {
            // issue next B stage into the other buffer (async, no VGPR data)
            const __bf16* g = bsrc + k0 + 32;
            __bf16* l = (buf ? bdst0 : bdst1);
#if defined(USE_ASYNC_LDS)
            async_cp16(g, l);
#else
            *(f32x4*)l = *(const f32x4*)g;
#endif
        }
#if defined(USE_ASYNC_LDS)
        // stage i complete when at most the just-issued stage i+1 is in flight
        if (i + 1 < nst) __builtin_amdgcn_s_wait_asynccnt(1);
        else             __builtin_amdgcn_s_wait_asynccnt(0);
#endif
        __syncthreads();

        // prefetch next A fragments into registers while WMMAs run
        if (i + 1 < nst) {
            const __bf16* pa = arow0 + k0 + 32 + half * 8;
            a0n.f[0] = *(const f32x4*)(pa);
            a0n.f[1] = *(const f32x4*)(pa + 16);
            pa = arow1 + k0 + 32 + half * 8;
            a1n.f[0] = *(const f32x4*)(pa);
            a1n.f[1] = *(const f32x4*)(pa + 16);
        }

        // Batch all B fragment loads (LDS returns in order -> partial waits),
        // then issue 8 independent WMMAs back-to-back.
        const __bf16* bb = &Bs[buf][0] + l16 * BROW + half * 8;
        FragBF b0, b1, b2, b3;
        b0.f[0] = *(const f32x4*)(bb + 0 * 16 * BROW);
        b0.f[1] = *(const f32x4*)(bb + 0 * 16 * BROW + 16);
        b1.f[0] = *(const f32x4*)(bb + 1 * 16 * BROW);
        b1.f[1] = *(const f32x4*)(bb + 1 * 16 * BROW + 16);
        b2.f[0] = *(const f32x4*)(bb + 2 * 16 * BROW);
        b2.f[1] = *(const f32x4*)(bb + 2 * 16 * BROW + 16);
        b3.f[0] = *(const f32x4*)(bb + 3 * 16 * BROW);
        b3.f[1] = *(const f32x4*)(bb + 3 * 16 * BROW + 16);

        acc00 = __builtin_amdgcn_wmma_f32_16x16x32_bf16(false, a0c.v, false, b0.v,
                                                        (short)0, acc00, false, false);
        acc01 = __builtin_amdgcn_wmma_f32_16x16x32_bf16(false, a0c.v, false, b1.v,
                                                        (short)0, acc01, false, false);
        acc02 = __builtin_amdgcn_wmma_f32_16x16x32_bf16(false, a0c.v, false, b2.v,
                                                        (short)0, acc02, false, false);
        acc03 = __builtin_amdgcn_wmma_f32_16x16x32_bf16(false, a0c.v, false, b3.v,
                                                        (short)0, acc03, false, false);
        acc10 = __builtin_amdgcn_wmma_f32_16x16x32_bf16(false, a1c.v, false, b0.v,
                                                        (short)0, acc10, false, false);
        acc11 = __builtin_amdgcn_wmma_f32_16x16x32_bf16(false, a1c.v, false, b1.v,
                                                        (short)0, acc11, false, false);
        acc12 = __builtin_amdgcn_wmma_f32_16x16x32_bf16(false, a1c.v, false, b2.v,
                                                        (short)0, acc12, false, false);
        acc13 = __builtin_amdgcn_wmma_f32_16x16x32_bf16(false, a1c.v, false, b3.v,
                                                        (short)0, acc13, false, false);

        a0c = a0n;
        a1c = a1n;
        __syncthreads();   // protect buf before next overwrite
    }

    if (!rowok) return;

    // D layout: lane -> N (=l16), VGPR v -> M = v + half*8
    float* crow0 = C + (size_t)(mtile * 32 + half * 8) * Fout + n0 + l16;
    float* crow1 = crow0 + (size_t)16 * Fout;
    #pragma unroll
    for (int v = 0; v < 8; ++v) {
        crow0[(size_t)v * Fout +  0] = acc00[v];
        crow0[(size_t)v * Fout + 16] = acc01[v];
        crow0[(size_t)v * Fout + 32] = acc02[v];
        crow0[(size_t)v * Fout + 48] = acc03[v];
        crow1[(size_t)v * Fout +  0] = acc10[v];
        crow1[(size_t)v * Fout + 16] = acc11[v];
        crow1[(size_t)v * Fout + 32] = acc12[v];
        crow1[(size_t)v * Fout + 48] = acc13[v];
    }
}

// =====================================================================
// Graph-structure kernels (computed once, reused by all 5 layers)
// =====================================================================
__global__ void k_set_deg(float* deg, int n) {
    int i = blockIdx.x * blockDim.x + threadIdx.x;
    if (i < n) deg[i] = 1.0f;                        // self-loop weight
}
__global__ void k_deg_edges(float* deg, const int* __restrict__ dst,
                            const float* __restrict__ ew, int E) {
    int e = blockIdx.x * blockDim.x + threadIdx.x;
    if (e < E) atomicAdd(&deg[dst[e]], ew[e]);
}
__global__ void k_dinv(float* dinv, int n) {
    int i = blockIdx.x * blockDim.x + threadIdx.x;
    if (i < n) { float d = dinv[i]; dinv[i] = (d > 0.f) ? rsqrtf(d) : 0.f; }
}
__global__ void k_norm(float* __restrict__ norm, const int* __restrict__ src,
                       const int* __restrict__ dst, const float* __restrict__ ew,
                       const float* __restrict__ dinv, int E) {
    int e = blockIdx.x * blockDim.x + threadIdx.x;
    if (e < E) norm[e] = dinv[src[e]] * ew[e] * dinv[dst[e]];
}

// =====================================================================
// Data-movement / elementwise kernels
// =====================================================================
__global__ void k_f32_to_bf16(const float* __restrict__ in, __bf16* __restrict__ out,
                              int count, int relu) {
    int i = blockIdx.x * blockDim.x + threadIdx.x;
    if (i < count) {
        float v = in[i];
        if (relu) v = fmaxf(v, 0.f);
        out[i] = (__bf16)v;
    }
}
__global__ void k_transpose_w(const float* __restrict__ W, __bf16* __restrict__ Wt,
                              int Fin, int Fout) {
    int i = blockIdx.x * blockDim.x + threadIdx.x;
    if (i < Fin * Fout) {
        int k = i / Fout, n = i % Fout;
        Wt[(size_t)n * Fin + k] = (__bf16)W[i];
    }
}

// OUT[n][f] = b[f] + dinv[n]^2 * H[n][f]       (self-loop + bias)
__global__ void k_agg_init(float* __restrict__ OUT, const float* __restrict__ H,
                           const float* __restrict__ b, const float* __restrict__ dinv,
                           int n, int F) {
    int fq = F >> 2;
    int t = blockIdx.x * blockDim.x + threadIdx.x;
    if (t >= n * fq) return;
    int node = t / fq, f4 = t % fq;
    float c = dinv[node]; c *= c;
    f32x4 h  = ((const f32x4*)(H + (size_t)node * F))[f4];
    f32x4 bv = ((const f32x4*)b)[f4];
    f32x4 o;
    o.x = bv.x + c * h.x;  o.y = bv.y + c * h.y;
    o.z = bv.z + c * h.z;  o.w = bv.w + c * h.w;
    ((f32x4*)(OUT + (size_t)node * F))[f4] = o;
}

// OUT[dst] += norm[e] * H[src]   (128 consecutive threads share one edge -> coalesced gather)
__global__ void k_agg_edges(float* __restrict__ OUT, const float* __restrict__ H,
                            const int* __restrict__ src, const int* __restrict__ dst,
                            const float* __restrict__ norm, int E, int F) {
    int fq = F >> 2;
    int t = blockIdx.x * blockDim.x + threadIdx.x;
    if (t >= E * fq) return;
    int e = t / fq, f4 = t % fq;
    int s = src[e], d = dst[e];
    float w = norm[e];
    f32x4 h = ((const f32x4*)(H + (size_t)s * F))[f4];
    float* o = OUT + (size_t)d * F + (size_t)f4 * 4;
    atomicAdd(o + 0, w * h.x);
    atomicAdd(o + 1, w * h.y);
    atomicAdd(o + 2, w * h.z);
    atomicAdd(o + 3, w * h.w);
}

// Layer 5: h5[i] = sum_k relu(X[i][k]) * w[k]   (one wave per node, K=256)
__global__ void k_dot5(const float* __restrict__ X, const float* __restrict__ w,
                       float* __restrict__ h5, int n, int K) {
    int gw = (blockIdx.x * blockDim.x + threadIdx.x) >> 5;
    int lane = threadIdx.x & 31;
    if (gw >= n) return;
    const float* row = X + (size_t)gw * K;
    float s = 0.f;
    for (int k = lane; k < K; k += 32) s += fmaxf(row[k], 0.f) * w[k];
    #pragma unroll
    for (int off = 16; off > 0; off >>= 1) s += __shfl_down(s, off, 32);
    if (lane == 0) h5[gw] = s;
}
__global__ void k_out_init(float* __restrict__ out, const float* __restrict__ h5,
                           const float* __restrict__ b5, const float* __restrict__ dinv, int n) {
    int i = blockIdx.x * blockDim.x + threadIdx.x;
    if (i < n) { float c = dinv[i]; out[i] = b5[0] + c * c * h5[i]; }
}
__global__ void k_out_edges(float* __restrict__ out, const float* __restrict__ h5,
                            const int* __restrict__ src, const int* __restrict__ dst,
                            const float* __restrict__ norm, int E) {
    int e = blockIdx.x * blockDim.x + threadIdx.x;
    if (e < E) atomicAdd(&out[dst[e]], norm[e] * h5[src[e]]);
}

// =====================================================================
// Launch
// =====================================================================
extern "C" void kernel_launch(void* const* d_in, const int* in_sizes, int n_in,
                              void* d_out, int out_size, void* d_ws, size_t ws_size,
                              hipStream_t stream) {
    const int IN_DIM = 512;
    const int N = in_sizes[0] / IN_DIM;      // 20000
    const int E = in_sizes[1] / 2;           // 400000

    const float* x   = (const float*)d_in[0];
    const int*   ei  = (const int*)d_in[1];
    const int*   src = ei;
    const int*   dst = ei + E;
    const float* ew  = (const float*)d_in[2];
    const float* Wp[5] = {(const float*)d_in[3], (const float*)d_in[5],
                          (const float*)d_in[7], (const float*)d_in[9],
                          (const float*)d_in[11]};
    const float* bp[5] = {(const float*)d_in[4], (const float*)d_in[6],
                          (const float*)d_in[8], (const float*)d_in[10],
                          (const float*)d_in[12]};
    const int dims[6] = {512, 512, 512, 512, 256, 1};

    // --- workspace carve-out ---
    size_t off = 0;
    auto carve = [&](size_t bytes) -> void* {
        void* p = (char*)d_ws + off;
        off += (bytes + 255) & ~(size_t)255;
        return p;
    };
    float*  dinv = (float*)carve((size_t)N * 4);
    float*  norm = (float*)carve((size_t)E * 4);
    float*  H    = (float*)carve((size_t)N * 512 * 4);
    float*  OUT  = (float*)carve((size_t)N * 512 * 4);
    __bf16* Xb   = (__bf16*)carve((size_t)N * 512 * 2);
    __bf16* Wt   = (__bf16*)carve((size_t)512 * 512 * 2);
    float*  h5   = (float*)carve((size_t)N * 4);
    if (off > ws_size) return;   // insufficient scratch

    const int B = 256;
    auto g1 = [](long long n, int b) { return (unsigned)((n + b - 1) / b); };

    // --- graph structure (once) ---
    k_set_deg  <<<g1(N, B), B, 0, stream>>>(dinv, N);
    k_deg_edges<<<g1(E, B), B, 0, stream>>>(dinv, dst, ew, E);
    k_dinv     <<<g1(N, B), B, 0, stream>>>(dinv, N);
    k_norm     <<<g1(E, B), B, 0, stream>>>(norm, src, dst, ew, dinv, E);

    // --- input features fp32 -> bf16 (no relu) ---
    k_f32_to_bf16<<<g1((long long)N * 512, B), B, 0, stream>>>(x, Xb, N * 512, 0);

    // --- layers 1..4: WMMA GEMM + aggregate (+ fused ReLU/convert) ---
    const int mtiles32 = N / 32;             // 625 (exact)
    for (int l = 0; l < 4; ++l) {
        const int Fin  = dims[l];            // 512
        const int Fout = dims[l + 1];        // 512/512/512/256

        k_transpose_w<<<g1((long long)Fin * Fout, B), B, 0, stream>>>(Wp[l], Wt, Fin, Fout);

        dim3 gg(g1(mtiles32, 8), Fout / 64);
        gemm_bf16_wmma<<<gg, B, 0, stream>>>(Xb, Wt, H, N, Fin, Fout);

        k_agg_init <<<g1((long long)N * (Fout / 4), B), B, 0, stream>>>(OUT, H, bp[l], dinv, N, Fout);
        k_agg_edges<<<g1((long long)E * (Fout / 4), B), B, 0, stream>>>(OUT, H, src, dst, norm, E, Fout);

        if (l < 3)   // next layer consumes bf16 relu(OUT)
            k_f32_to_bf16<<<g1((long long)N * Fout, B), B, 0, stream>>>(OUT, Xb, N * Fout, 1);
    }

    // --- layer 5 (Fout = 1): wave-reduction dot + scalar aggregation ---
    k_dot5    <<<g1((long long)N * 32, B), B, 0, stream>>>(OUT, Wp[4], h5, N, 256);
    k_out_init<<<g1(N, B), B, 0, stream>>>((float*)d_out, h5, bp[4], dinv, N);
    k_out_edges<<<g1(E, B), B, 0, stream>>>((float*)d_out, h5, src, dst, norm, E);
}
